// HierarchicalReasoningModel_ACTV1Block_85478439125246
// MI455X (gfx1250) — matrix operand order repaired and checked by reading, not compiled, not run on hardware
//
#include <hip/hip_runtime.h>

#define T_TOK 1024
#define H_DIM 1024
#define E_ROUT 16
#define IR 1024
#define NSH 2
#define IS 2816
#define BLD 40   // transposed-B LDS leading dim (bf16 units): 80B, keeps 16B alignment

typedef __attribute__((ext_vector_type(16))) __bf16 bf16x16;
typedef __attribute__((ext_vector_type(8)))  __bf16 bf16x8;
typedef __attribute__((ext_vector_type(4)))  __bf16 bf16x4;
typedef __attribute__((ext_vector_type(8)))  float  f32x8;

__device__ __forceinline__ __bf16 f2bf(float x) { return (__bf16)x; }

// A fragment 16x32 (MxK) bf16 from LDS tile (row-major, leading dim ldk)
// lane L: row = L%16 ; K chunks [8h..8h+7] and [16+8h..16+8h+7], h = L/16
__device__ __forceinline__ bf16x16 lds_load_a16x32(const __bf16* base, int row0, int ldk) {
  const int lane = threadIdx.x & 31;
  const int hh = lane >> 4;
  const __bf16* rp = base + (row0 + (lane & 15)) * ldk + 8 * hh;
  bf16x16 f;
#pragma unroll
  for (int t = 0; t < 8; ++t) f[t] = rp[t];
#pragma unroll
  for (int t = 0; t < 8; ++t) f[8 + t] = rp[16 + t];
  return f;
}

// B fragment 32x16 (KxN) bf16 from LDS tile stored TRANSPOSED: [n][k], ld=BLD
// lane L: col = L%16 ; K chunks [8h..8h+7] and [16+8h..16+8h+7], h = L/16 (2x b128)
__device__ __forceinline__ bf16x16 lds_load_b32x16t(const __bf16* baseT, int col0) {
  const int lane = threadIdx.x & 31;
  const int hh = lane >> 4;
  const __bf16* cp = baseT + (col0 + (lane & 15)) * BLD + 8 * hh;
  bf16x16 f;
#pragma unroll
  for (int t = 0; t < 8; ++t) f[t] = cp[t];
#pragma unroll
  for (int t = 0; t < 8; ++t) f[8 + t] = cp[16 + t];
  return f;
}

__global__ void init_aux_kernel(float* aux) {
  int i = threadIdx.x;
  if (i < 40) aux[i] = 0.0f;
}

__global__ __launch_bounds__(256)
void convert_x_kernel(const float* __restrict__ x, __bf16* __restrict__ xbf) {
  int i = (blockIdx.x * 256 + threadIdx.x) * 4;
  float4 v = *(const float4*)&x[i];
  bf16x4 o;
  o[0] = f2bf(v.x); o[1] = f2bf(v.y); o[2] = f2bf(v.z); o[3] = f2bf(v.w);
  *(bf16x4*)&xbf[i] = o;
}

__global__ __launch_bounds__(256)
void router_kernel(const float* __restrict__ x, const float* __restrict__ rw,
                   float* __restrict__ C, float* __restrict__ aux) {
  int t = blockIdx.x * blockDim.x + threadIdx.x;
  if (t >= T_TOK) return;
  float logit[16];
#pragma unroll
  for (int e = 0; e < 16; ++e) logit[e] = 0.0f;
  const float* xr = x + t * H_DIM;
  for (int h = 0; h < H_DIM; ++h) {
    float xv = xr[h];
#pragma unroll
    for (int e = 0; e < 16; ++e) logit[e] += xv * rw[e * H_DIM + h];
  }
  float mx = -1e30f;
#pragma unroll
  for (int e = 0; e < 16; ++e) {
    float l = fminf(20.0f, fmaxf(-20.0f, logit[e]));
    logit[e] = l; mx = fmaxf(mx, l);
  }
  float s[16]; float sum = 0.0f;
#pragma unroll
  for (int e = 0; e < 16; ++e) { s[e] = __expf(logit[e] - mx); sum += s[e]; }
  float inv = 1.0f / sum;
#pragma unroll
  for (int e = 0; e < 16; ++e) s[e] *= inv;

  // device scores (e2d = e % 8)
  float ds[8];
#pragma unroll
  for (int d = 0; d < 8; ++d) ds[d] = s[d] + s[d + 8];
  // top-3 devices
  int selmask = 0;
  for (int k = 0; k < 3; ++k) {
    int best = 0; float bv = -1e30f;
    for (int d = 0; d < 8; ++d)
      if (!((selmask >> d) & 1) && ds[d] > bv) { bv = ds[d]; best = d; }
    selmask |= (1 << best);
  }
  float m[16];
#pragma unroll
  for (int e = 0; e < 16; ++e) m[e] = ((selmask >> (e & 7)) & 1) ? s[e] : -1e30f;
  // top-6 experts among masked
  int idx[6]; float raw[6];
  for (int k = 0; k < 6; ++k) {
    int best = 0; float bv = -2e30f;
    for (int e = 0; e < 16; ++e)
      if (m[e] > bv) { bv = m[e]; best = e; }
    idx[k] = best; raw[k] = bv; m[best] = -3e30f;
  }
  // renormalize (stable softmax over k; raw[0] is max)
  float rmx = raw[0];
  float wv[6]; float wsum = 0.0f;
  for (int k = 0; k < 6; ++k) { wv[k] = __expf(raw[k] - rmx); wsum += wv[k]; }
  float winv = 1.0f / (wsum + 1e-9f);
  float crow[16];
#pragma unroll
  for (int e = 0; e < 16; ++e) crow[e] = 0.0f;
  for (int k = 0; k < 6; ++k) crow[idx[k]] += wv[k] * winv;
#pragma unroll
  for (int e = 0; e < 16; ++e) C[t * 16 + e] = crow[e];

  // aux stats: [0..15] select counts, [16..31] score sums, [32..39] device counts
  for (int k = 0; k < 6; ++k) {
    atomicAdd(&aux[idx[k]], 1.0f);
    atomicAdd(&aux[32 + (idx[k] & 7)], 1.0f);
  }
#pragma unroll
  for (int e = 0; e < 16; ++e) atomicAdd(&aux[16 + e], s[e]);
}

__global__ void aux_finalize_kernel(const float* __restrict__ aux, float* __restrict__ out_aux) {
  if (threadIdx.x != 0 || blockIdx.x != 0) return;
  float f[16], P[16];
  for (int e = 0; e < 16; ++e) {
    f[e] = aux[e] / (float)(T_TOK * 6);
    P[e] = aux[16 + e] / (float)T_TOK;
  }
  float eb = 0.0f;
  for (int e = 0; e < 16; ++e) eb += f[e] * P[e];
  eb *= 0.003f;
  float db = 0.0f, cb = 0.0f;
  for (int d = 0; d < 8; ++d) {
    float df = 0.5f * (f[d] + f[d + 8]);
    float dP = P[d] + P[d + 8];
    db += df * dP;
    float fc = aux[32 + d] / (float)(T_TOK * 3);
    cb += fc * dP;
  }
  db *= 0.05f; cb *= 0.02f;
  *out_aux = eb + db + cb;
}

// Stage A tile (64x32 bf16, row-major) with CDNA5 async global->LDS copies.
// Each thread issues two GLOBAL_LOAD_ASYNC_TO_LDS_B128 (no VGPR round-trip,
// tracked by ASYNCcnt). Caller must s_wait_asynccnt before the barrier.
__device__ __forceinline__ void stage_a_async(__bf16* lsA, const __bf16* src, int t0,
                                              int lda, int kk, int tid) {
#pragma unroll
  for (int i = tid; i < 64 * 32 / 8; i += 128) {   // 256 groups of 8 bf16 (16B)
    int r = i >> 2, c8 = (i & 3) * 8;
    const __bf16* gp = &src[(size_t)(t0 + r) * lda + kk + c8];
    unsigned lp = (unsigned)(size_t)&lsA[r * 32 + c8];   // low 32b = LDS offset
    asm volatile("global_load_async_to_lds_b128 %0, %1, off"
                 :: "v"(lp), "v"((unsigned long long)(size_t)gp)
                 : "memory");
  }
}

__device__ __forceinline__ void wait_async_all() {
  asm volatile("s_wait_asynccnt 0x0" ::: "memory");
}

// Stage B tile 32x64 (KxN) -> transposed LDS [n][k]: each thread takes 4
// consecutive K at fixed n (coalesced in n across lanes), packed b64 store.
__device__ __forceinline__ void stage_bT(__bf16* lsBT, const float* W, int kk,
                                         int n0, int ld, int tid) {
#pragma unroll
  for (int i = tid; i < 512; i += 128) {           // 64 n x 8 k-groups
    int n = i & 63, g = i >> 6;
    const float* p = W + (size_t)(kk + g * 4) * ld + n0 + n;
    bf16x4 o;
    o[0] = f2bf(p[0]);
    o[1] = f2bf(p[(size_t)ld]);
    o[2] = f2bf(p[(size_t)2 * ld]);
    o[3] = f2bf(p[(size_t)3 * ld]);
    *(bf16x4*)&lsBT[n * BLD + g * 4] = o;
  }
}

// Fused gate+up GEMM + SwiGLU + combine-weight, per expert (blockIdx.z).
// Out tile: 64 tokens x 64 intermediate cols. 4 waves, each 32x32 (2x2 WMMA).
__global__ __launch_bounds__(128)
void gateup_kernel(const __bf16* __restrict__ xbf,
                   const float* __restrict__ Wg,
                   const float* __restrict__ Wu,
                   const float* __restrict__ Cmat,
                   __bf16* __restrict__ act,
                   int I, int actStride, float constScale, int useC) {
  __shared__ alignas(16) __bf16 lsA[64 * 32];
  __shared__ alignas(16) __bf16 lsBgT[64 * BLD];
  __shared__ alignas(16) __bf16 lsBuT[64 * BLD];

  const int tid  = threadIdx.x;
  const int wave = tid >> 5;
  const int n0 = blockIdx.x * 64;
  const int t0 = blockIdx.y * 64;
  const int e  = blockIdx.z;
  const int wm = (wave >> 1) * 32;
  const int wn = (wave & 1) * 32;
  const float* wgE = Wg + (size_t)e * H_DIM * I;
  const float* wuE = Wu + (size_t)e * H_DIM * I;

  f32x8 zero = {0.f, 0.f, 0.f, 0.f, 0.f, 0.f, 0.f, 0.f};
  f32x8 accg[2][2], accu[2][2];
#pragma unroll
  for (int a = 0; a < 2; ++a)
#pragma unroll
    for (int b = 0; b < 2; ++b) { accg[a][b] = zero; accu[a][b] = zero; }

  for (int kk = 0; kk < H_DIM; kk += 32) {
    stage_a_async(lsA, xbf, t0, H_DIM, kk, tid);   // async mem->LDS, no VGPRs
    stage_bT(lsBgT, wgE, kk, n0, I, tid);          // overlaps with async copy
    stage_bT(lsBuT, wuE, kk, n0, I, tid);
    wait_async_all();
    __syncthreads();
    bf16x16 a0  = lds_load_a16x32(lsA, wm, 32);
    bf16x16 a1  = lds_load_a16x32(lsA, wm + 16, 32);
    bf16x16 bg0 = lds_load_b32x16t(lsBgT, wn);
    bf16x16 bg1 = lds_load_b32x16t(lsBgT, wn + 16);
    bf16x16 bu0 = lds_load_b32x16t(lsBuT, wn);
    bf16x16 bu1 = lds_load_b32x16t(lsBuT, wn + 16);
    accg[0][0] = __builtin_amdgcn_wmma_f32_16x16x32_bf16(false, a0, false, bg0, (short)0, accg[0][0], false, false);
    accg[0][1] = __builtin_amdgcn_wmma_f32_16x16x32_bf16(false, a0, false, bg1, (short)0, accg[0][1], false, false);
    accg[1][0] = __builtin_amdgcn_wmma_f32_16x16x32_bf16(false, a1, false, bg0, (short)0, accg[1][0], false, false);
    accg[1][1] = __builtin_amdgcn_wmma_f32_16x16x32_bf16(false, a1, false, bg1, (short)0, accg[1][1], false, false);
    accu[0][0] = __builtin_amdgcn_wmma_f32_16x16x32_bf16(false, a0, false, bu0, (short)0, accu[0][0], false, false);
    accu[0][1] = __builtin_amdgcn_wmma_f32_16x16x32_bf16(false, a0, false, bu1, (short)0, accu[0][1], false, false);
    accu[1][0] = __builtin_amdgcn_wmma_f32_16x16x32_bf16(false, a1, false, bu0, (short)0, accu[1][0], false, false);
    accu[1][1] = __builtin_amdgcn_wmma_f32_16x16x32_bf16(false, a1, false, bu1, (short)0, accu[1][1], false, false);
    __syncthreads();
  }

  const int lane = tid & 31;
  const int hh = lane >> 4;
  const int nl = lane & 15;
#pragma unroll
  for (int mi = 0; mi < 2; ++mi)
#pragma unroll
    for (int ni = 0; ni < 2; ++ni)
#pragma unroll
      for (int r = 0; r < 8; ++r) {
        int row = t0 + wm + mi * 16 + hh * 8 + r;
        int col = n0 + wn + ni * 16 + nl;
        float g = accg[mi][ni][r];
        float u = accu[mi][ni][r];
        // silu(g)*u with fast reciprocal (v_rcp_f32)
        float a = g * __builtin_amdgcn_rcpf(1.0f + __expf(-g)) * u;
        float w = useC ? Cmat[row * E_ROUT + e] : constScale;
        act[(size_t)row * actStride + e * I + col] = f2bf(a * w);
      }
}

// Down projection: out[T,H] (+)= act[T,Ktot] @ W[Ktot,H]
__global__ __launch_bounds__(128)
void down_kernel(const __bf16* __restrict__ act,
                 const float* __restrict__ W,
                 float* __restrict__ out, int Ktot, int accumulate) {
  __shared__ alignas(16) __bf16 lsA[64 * 32];
  __shared__ alignas(16) __bf16 lsBT[64 * BLD];
  const int tid  = threadIdx.x;
  const int wave = tid >> 5;
  const int n0 = blockIdx.x * 64;
  const int t0 = blockIdx.y * 64;
  const int wm = (wave >> 1) * 32;
  const int wn = (wave & 1) * 32;

  f32x8 zero = {0.f, 0.f, 0.f, 0.f, 0.f, 0.f, 0.f, 0.f};
  f32x8 acc[2][2];
#pragma unroll
  for (int a = 0; a < 2; ++a)
#pragma unroll
    for (int b = 0; b < 2; ++b) acc[a][b] = zero;

  for (int kk = 0; kk < Ktot; kk += 32) {
    stage_a_async(lsA, act, t0, Ktot, kk, tid);
    stage_bT(lsBT, W, kk, n0, H_DIM, tid);
    wait_async_all();
    __syncthreads();
    bf16x16 a0 = lds_load_a16x32(lsA, wm, 32);
    bf16x16 a1 = lds_load_a16x32(lsA, wm + 16, 32);
    bf16x16 b0 = lds_load_b32x16t(lsBT, wn);
    bf16x16 b1 = lds_load_b32x16t(lsBT, wn + 16);
    acc[0][0] = __builtin_amdgcn_wmma_f32_16x16x32_bf16(false, a0, false, b0, (short)0, acc[0][0], false, false);
    acc[0][1] = __builtin_amdgcn_wmma_f32_16x16x32_bf16(false, a0, false, b1, (short)0, acc[0][1], false, false);
    acc[1][0] = __builtin_amdgcn_wmma_f32_16x16x32_bf16(false, a1, false, b0, (short)0, acc[1][0], false, false);
    acc[1][1] = __builtin_amdgcn_wmma_f32_16x16x32_bf16(false, a1, false, b1, (short)0, acc[1][1], false, false);
    __syncthreads();
  }

  const int lane = tid & 31;
  const int hh = lane >> 4;
  const int nl = lane & 15;
#pragma unroll
  for (int mi = 0; mi < 2; ++mi)
#pragma unroll
    for (int ni = 0; ni < 2; ++ni)
#pragma unroll
      for (int r = 0; r < 8; ++r) {
        int row = t0 + wm + mi * 16 + hh * 8 + r;
        int col = n0 + wn + ni * 16 + nl;
        size_t o = (size_t)row * H_DIM + col;
        float v = acc[mi][ni][r];
        out[o] = accumulate ? (out[o] + v) : v;
      }
}

extern "C" void kernel_launch(void* const* d_in, const int* in_sizes, int n_in,
                              void* d_out, int out_size, void* d_ws, size_t ws_size,
                              hipStream_t stream) {
  const float* x      = (const float*)d_in[0];
  const float* rw     = (const float*)d_in[1];
  const float* r_gate = (const float*)d_in[2];
  const float* r_up   = (const float*)d_in[3];
  const float* r_down = (const float*)d_in[4];
  const float* s_gate = (const float*)d_in[5];
  const float* s_up   = (const float*)d_in[6];
  const float* s_down = (const float*)d_in[7];
  float* out = (float*)d_out;

  char* ws = (char*)d_ws;
  __bf16* xbf  = (__bf16*)(ws);                                                  // 2 MB
  float*  C    = (float*)(ws + (size_t)2 * 1024 * 1024);                         // 64 KB
  float*  aux  = (float*)(ws + (size_t)2 * 1024 * 1024 + 65536);                 // 160 B
  __bf16* actR = (__bf16*)(ws + (size_t)4 * 1024 * 1024);                        // 32 MB
  __bf16* actS = (__bf16*)(ws + (size_t)4 * 1024 * 1024 +
                           (size_t)T_TOK * E_ROUT * IR * 2);                     // 11 MB

  init_aux_kernel<<<1, 64, 0, stream>>>(aux);
  convert_x_kernel<<<(T_TOK * H_DIM) / 1024, 256, 0, stream>>>(x, xbf);
  router_kernel<<<T_TOK / 256, 256, 0, stream>>>(x, rw, C, aux);
  aux_finalize_kernel<<<1, 1, 0, stream>>>(aux, out + (size_t)T_TOK * H_DIM);

  gateup_kernel<<<dim3(IR / 64, T_TOK / 64, E_ROUT), 128, 0, stream>>>(
      xbf, r_gate, r_up, C, actR, IR, E_ROUT * IR, 0.0f, 1);
  gateup_kernel<<<dim3(IS / 64, T_TOK / 64, NSH), 128, 0, stream>>>(
      xbf, s_gate, s_up, nullptr, actS, IS, NSH * IS, 0.5f, 0);

  down_kernel<<<dim3(H_DIM / 64, T_TOK / 64), 128, 0, stream>>>(
      actS, s_down, out, NSH * IS, 0);
  down_kernel<<<dim3(H_DIM / 64, T_TOK / 64), 128, 0, stream>>>(
      actR, r_down, out, E_ROUT * IR, 1);
}
